// MHA_QAgent_88768384074448
// MI455X (gfx1250) — matrix operand order repaired and checked by reading, not compiled, not run on hardware
//
#include <hip/hip_runtime.h>
#include <hip/hip_bf16.h>

typedef __attribute__((ext_vector_type(16))) __bf16 v16bf;
typedef __attribute__((ext_vector_type(8)))  __bf16 v8bf;
typedef __attribute__((ext_vector_type(8)))  float  v8f;

#define WMMA_BF16(a, b, c) \
  __builtin_amdgcn_wmma_f32_16x16x32_bf16(false, (a), false, (b), (short)0, (c), false, false)

// ---------------------------------------------------------------------------
// Vectorized fragment loaders (no guards; K multiple of 32, aligned buffers)
//
// A (16x32 bf16): lanes 0-15 -> M=lane,   K = kb + {0..7, 16..23}
//                 lanes16-31 -> M=lane-16,K = kb + {8..15,24..31}
// => two contiguous 16B loads at rowPtr + kb + half*8 (+16)
// ---------------------------------------------------------------------------
__device__ __forceinline__ v16bf load_afrag(const __bf16* rowPtr, int kb) {
  int half = (threadIdx.x >> 4) & 1;
  const __bf16* p = rowPtr + kb + half * 8;
  v8bf lo = *(const v8bf*)(p);
  v8bf hi = *(const v8bf*)(p + 16);
  v16bf f;
#pragma unroll
  for (int e = 0; e < 8; ++e) { f[e] = lo[e]; f[8 + e] = hi[e]; }
  return f;
}

// B fragments come from a pre-swizzled weight layout:
// block (bk,bn) of 32k x 16n => 512 contiguous bf16; within a block each lane
// owns 16 contiguous bf16 (its full fragment) => one aligned 32B load.
__device__ __forceinline__ v16bf load_bfrag(const __bf16* Bsw, int N, int kb, int nb) {
  long block = (long)(kb >> 5) * (N >> 4) + (nb >> 4);
  const __bf16* p = Bsw + block * 512 + (threadIdx.x & 31) * 16;
  return *(const v16bf*)p;
}

// ---------------------------------------------------------------------------
// Weight swizzle: src f32 [Ksrc,N] (trans=0) or [N,Ksrc] (trans=1)
// -> dst bf16 swizzled, K padded to multiple of 32 with zeros.
// ---------------------------------------------------------------------------
__global__ void swizzleB_kernel(const float* __restrict__ src, __bf16* __restrict__ dst,
                                int K, int Ksrc, int N, int trans) {
  long i = (long)blockIdx.x * blockDim.x + threadIdx.x;
  long total = (long)K * N;
  if (i >= total) return;
  long block = i >> 9;
  int rem = (int)(i & 511);
  int lane = rem >> 4;
  int e = rem & 15;
  int nbl = N >> 4;
  int bk = (int)(block / nbl), bn = (int)(block % nbl);
  int k = bk * 32 + ((e >> 3) << 4) + ((lane >> 4) << 3) + (e & 7);
  int n = bn * 16 + (lane & 15);
  float v = 0.f;
  if (k < Ksrc) v = trans ? src[(long)n * Ksrc + k] : src[(long)k * N + n];
  dst[i] = (__bf16)v;
}

// ---------------------------------------------------------------------------
// Generic bf16 GEMM: C[M,N] = A[M,K] @ Bsw + bias.
// Block 256 thr = 8 waves; block tile 256x64; wave tile 32x64
// (2 A-frags share 4 B-frags -> 8 WMMA / k-step).
// Ping-pong double buffer (no register moves): buffer sets A/B alternate.
// Optional f32 / bf16 outputs with ldc + column offset (free concat).
// ---------------------------------------------------------------------------
__global__ __launch_bounds__(256) void gemm_bf16_kernel(
    const __bf16* __restrict__ A, int lda,
    const __bf16* __restrict__ Bsw,
    const float* __restrict__ bias,
    float* __restrict__ Cf, __bf16* __restrict__ Cb, int ldc, int ccoff,
    int M, int N, int K) {
  int wave = threadIdx.x >> 5;
  int lane = threadIdx.x & 31;
  int half = (lane >> 4) & 1;
  int mrow = blockIdx.y * 256 + wave * 32;
  int ncol = blockIdx.x * 64;
  if (mrow >= M) return;

  v8f acc[2][4] = {};
  const __bf16* rowPtr0 = A + (long)(mrow + (lane & 15)) * lda;
  const __bf16* rowPtr1 = A + (long)(mrow + 16 + (lane & 15)) * lda;
  int ksteps = K >> 5;  // >= 3 for all our shapes

  v16bf a0A, a1A, bA[4];   // buffer set A
  v16bf a0B, a1B, bB[4];   // buffer set B

  auto loadSet = [&](v16bf& x0, v16bf& x1, v16bf (&bb)[4], int kb) {
    x0 = load_afrag(rowPtr0, kb);
    x1 = load_afrag(rowPtr1, kb);
#pragma unroll
    for (int f = 0; f < 4; ++f) bb[f] = load_bfrag(Bsw, N, kb, ncol + 16 * f);
  };
  auto compSet = [&](const v16bf& x0, const v16bf& x1, const v16bf (&bb)[4]) {
#pragma unroll
    for (int f = 0; f < 4; ++f) {
      acc[0][f] = WMMA_BF16(x0, bb[f], acc[0][f]);
      acc[1][f] = WMMA_BF16(x1, bb[f], acc[1][f]);
    }
  };

  loadSet(a0A, a1A, bA, 0);
  int ks = 0;
  for (; ks + 2 < ksteps; ks += 2) {
    loadSet(a0B, a1B, bB, (ks + 1) * 32);
    compSet(a0A, a1A, bA);
    loadSet(a0A, a1A, bA, (ks + 2) * 32);
    compSet(a0B, a1B, bB);
  }
  if (ks + 1 < ksteps) {          // two steps remain
    loadSet(a0B, a1B, bB, (ks + 1) * 32);
    compSet(a0A, a1A, bA);
    compSet(a0B, a1B, bB);
  } else {                        // one step remains
    compSet(a0A, a1A, bA);
  }

#pragma unroll
  for (int t = 0; t < 2; ++t) {
#pragma unroll
    for (int f = 0; f < 4; ++f) {
      int n = ncol + 16 * f + (lane & 15);
      float bv = bias ? bias[n] : 0.0f;
#pragma unroll
      for (int r = 0; r < 8; ++r) {
        int mm = mrow + t * 16 + half * 8 + r;
        float v = acc[t][f][r] + bv;
        long idx = (long)mm * ldc + ccoff + n;
        if (Cf) Cf[idx] = v;
        if (Cb) Cb[idx] = (__bf16)v;
      }
    }
  }
}

// ---------------------------------------------------------------------------
// Fused entity attention, one block per batch element, 8 waves = 8 heads:
//   emb[16,512] = ent_pad[16,96] @ Wemb + bemb   (WMMA -> LDS bf16)
//   K,V[16,64]  = emb @ Wk/Wv  (head slice, accumulators stay in registers)
//   masked softmax via wave32 shfl reductions, out = p @ V -> bf16
// ---------------------------------------------------------------------------
__global__ __launch_bounds__(256) void attn_fused_kernel(
    const __bf16* __restrict__ ent,    // [B*16, 96] padded
    const __bf16* __restrict__ WembSw, // swizzled [96, 512]
    const float*  __restrict__ bemb,   // [512]
    const __bf16* __restrict__ WkSw,   // swizzled [512, 512]
    const __bf16* __restrict__ WvSw,   // swizzled [512, 512]
    const float*  __restrict__ qvec,   // [B, 512] f32
    const float*  __restrict__ mask,   // [B, 16]  (1 valid / 0 invalid)
    __bf16* __restrict__ outp)         // [B, 512] bf16
{
  __shared__ __bf16 Aloc[16 * 512];
  int b = blockIdx.x;
  int wave = threadIdx.x >> 5;   // head index
  int lane = threadIdx.x & 31;
  int half = (lane >> 4) & 1;
  int ncol = wave * 64;

  // Phase 1: embedding tile -> LDS
  {
    v8f acc[4] = {};
    const __bf16* rowPtr = ent + (long)(b * 16 + (lane & 15)) * 96;
    for (int ks = 0; ks < 3; ++ks) {
      int kb = ks * 32;
      v16bf a = load_afrag(rowPtr, kb);
      v16bf bb[4];
#pragma unroll
      for (int f = 0; f < 4; ++f) bb[f] = load_bfrag(WembSw, 512, kb, ncol + 16 * f);
#pragma unroll
      for (int f = 0; f < 4; ++f) acc[f] = WMMA_BF16(a, bb[f], acc[f]);
    }
#pragma unroll
    for (int f = 0; f < 4; ++f) {
      int n = ncol + 16 * f + (lane & 15);
      float bv = bemb[n];
#pragma unroll
      for (int r = 0; r < 8; ++r)
        Aloc[(half * 8 + r) * 512 + n] = (__bf16)(acc[f][r] + bv);
    }
  }
  __syncthreads();

  // Phase 2: K,V head slices (all loads clause-issued before the 8 WMMAs)
  v8f kacc[4] = {};
  v8f vacc[4] = {};
  {
    const __bf16* rowPtr = Aloc + (long)(lane & 15) * 512;
    for (int ks = 0; ks < 16; ++ks) {
      int kb = ks * 32;
      v16bf a = load_afrag(rowPtr, kb);
      v16bf bk[4], bv[4];
#pragma unroll
      for (int f = 0; f < 4; ++f) {
        bk[f] = load_bfrag(WkSw, 512, kb, ncol + 16 * f);
        bv[f] = load_bfrag(WvSw, 512, kb, ncol + 16 * f);
      }
#pragma unroll
      for (int f = 0; f < 4; ++f) {
        kacc[f] = WMMA_BF16(a, bk[f], kacc[f]);
        vacc[f] = WMMA_BF16(a, bv[f], vacc[f]);
      }
    }
  }

  // Phase 3: scores + masked softmax over 16 (padded) keys
  float qv[4];
#pragma unroll
  for (int f = 0; f < 4; ++f)
    qv[f] = qvec[(long)b * 512 + ncol + 16 * f + (lane & 15)];

  float p[16];
#pragma unroll
  for (int r = 0; r < 8; ++r) {
    float s = 0.f;
#pragma unroll
    for (int f = 0; f < 4; ++f) s += qv[f] * kacc[f][r];
    s += __shfl_xor(s, 1);
    s += __shfl_xor(s, 2);
    s += __shfl_xor(s, 4);
    s += __shfl_xor(s, 8);          // sum over the 16 lanes of this half
    float so = __shfl_xor(s, 16);   // other half's row set
    p[half ? 8 + r : r] = s;
    p[half ? r : 8 + r] = so;
  }
  float mx = -1e30f;
#pragma unroll
  for (int l = 0; l < 16; ++l) {
    float mk = mask[(long)b * 16 + l];
    p[l] = (mk != 0.0f) ? p[l] * 0.125f : -1e9f;   // 1/sqrt(64)
    mx = fmaxf(mx, p[l]);
  }
  float sum = 0.f;
#pragma unroll
  for (int l = 0; l < 16; ++l) { p[l] = __expf(p[l] - mx); sum += p[l]; }
  float inv = 1.0f / sum;
#pragma unroll
  for (int l = 0; l < 16; ++l) p[l] *= inv;

  // Phase 4: out[d] = sum_l p_l * V[l,d]
#pragma unroll
  for (int f = 0; f < 4; ++f) {
    float o = 0.f;
#pragma unroll
    for (int r = 0; r < 8; ++r) o += p[half * 8 + r] * vacc[f][r];
    o += __shfl_xor(o, 16);
    if (half == 0)
      outp[(long)b * 512 + ncol + 16 * f + lane] = (__bf16)o;
  }
}

// ---------------------------------------------------------------------------
// Helper kernels
// ---------------------------------------------------------------------------
__global__ void f2bf_kernel(const float* __restrict__ src, __bf16* __restrict__ dst, long n) {
  long i = (long)blockIdx.x * blockDim.x + threadIdx.x;
  long stride = (long)gridDim.x * blockDim.x;
  for (; i < n; i += stride) dst[i] = (__bf16)src[i];
}

// Pad entity [B*L, D] f32 -> [B*16, 96] bf16 (zero rows l>=L, cols c>=D)
template <int L, int D>
__global__ void pad_ent_kernel(const float* __restrict__ src, __bf16* __restrict__ dst, int B) {
  long i = (long)blockIdx.x * blockDim.x + threadIdx.x;
  long total = (long)B * 16 * 96;
  if (i >= total) return;
  long row = i / 96;
  int c = (int)(i % 96);
  long b = row >> 4;
  int l = (int)(row & 15);
  float v = 0.f;
  if (l < L && c < D) v = src[((long)b * L + l) * D + c];
  dst[i] = (__bf16)v;
}

template <int L, int D>
__global__ void mask_kernel(const float* __restrict__ ent, float* __restrict__ mask, int B) {
  int idx = blockIdx.x * blockDim.x + threadIdx.x;
  int b = idx >> 4, l = idx & 15;
  if (b >= B) return;
  float m = 0.f;
  if (l < L) {
    const float* p = ent + ((long)b * L + l) * D;
    bool allz = true;
    for (int d = 0; d < D; ++d)
      if (p[d] != 0.f) { allz = false; break; }
    m = allz ? 0.f : 1.f;
  }
  mask[idx] = m;
}

__global__ void gru_kernel(const float* __restrict__ gi, const float* __restrict__ gh,
                           const float* __restrict__ prevh, float* __restrict__ h, int B) {
  long idx = (long)blockIdx.x * blockDim.x + threadIdx.x;
  if (idx >= (long)B * 512) return;
  long b = idx >> 9;
  int j = (int)(idx & 511);
  const float* gib = gi + b * 1536;
  const float* ghb = gh + b * 1536;
  float r = 1.f / (1.f + __expf(-(gib[j]       + ghb[j])));
  float z = 1.f / (1.f + __expf(-(gib[512 + j] + ghb[512 + j])));
  float n = tanhf(gib[1024 + j] + r * ghb[1024 + j]);
  h[idx] = (1.f - z) * n + z * prevh[idx];
}

__global__ void qout_kernel(const float* __restrict__ h, const float* __restrict__ Wq,
                            const float* __restrict__ bq, float* __restrict__ out, int B) {
  int idx = blockIdx.x * blockDim.x + threadIdx.x;
  int b = idx / 22, o = idx % 22;
  if (b >= B) return;
  const float* hb = h + (long)b * 512;
  float s = bq[o];
  for (int k = 0; k < 512; ++k) s += hb[k] * Wq[k * 22 + o];
  out[(long)b * 22 + o] = s;
}

// ---------------------------------------------------------------------------
extern "C" void kernel_launch(void* const* d_in, const int* in_sizes, int n_in,
                              void* d_out, int out_size, void* d_ws, size_t ws_size,
                              hipStream_t stream) {
  (void)in_sizes; (void)n_in; (void)out_size; (void)ws_size;
  constexpr int B = 8192, NA = 15, NE = 16, DO = 96, DA = 88, DE = 80, H = 512;

  const float* own    = (const float*)d_in[0];
  const float* ally   = (const float*)d_in[1];
  const float* enemy  = (const float*)d_in[2];
  const float* prevh  = (const float*)d_in[3];
  const float* W_own  = (const float*)d_in[4];
  const float* b_own  = (const float*)d_in[5];
  const float* W_ally = (const float*)d_in[6];
  const float* b_ally = (const float*)d_in[7];
  const float* W_en   = (const float*)d_in[8];
  const float* b_en   = (const float*)d_in[9];
  const float* Wq_a = (const float*)d_in[10];
  const float* Wk_a = (const float*)d_in[11];
  const float* Wv_a = (const float*)d_in[12];
  const float* Wo_a = (const float*)d_in[13];
  const float* Wq_e = (const float*)d_in[14];
  const float* Wk_e = (const float*)d_in[15];
  const float* Wv_e = (const float*)d_in[16];
  const float* Wo_e = (const float*)d_in[17];
  const float* W_ih = (const float*)d_in[18];
  const float* W_hh = (const float*)d_in[19];
  const float* b_ih = (const float*)d_in[20];
  const float* b_hh = (const float*)d_in[21];
  const float* W_q  = (const float*)d_in[22];
  const float* b_q  = (const float*)d_in[23];
  float* outq = (float*)d_out;

  // --- workspace carve-up -------------------------------------------------
  char* wp = (char*)d_ws;
  auto carve = [&](size_t bytes) -> char* {
    char* p = wp;
    wp += (bytes + 255) & ~(size_t)255;
    return p;
  };
  __bf16* own16   = (__bf16*)carve((size_t)B * DO * 2);
  __bf16* allyP   = (__bf16*)carve((size_t)B * 16 * 96 * 2);  // padded
  __bf16* enemyP  = (__bf16*)carve((size_t)B * 16 * 96 * 2);  // padded
  __bf16* prevh16 = (__bf16*)carve((size_t)B * H * 2);
  __bf16* wOwnSw  = (__bf16*)carve((size_t)DO * H * 2);        // K=96
  __bf16* wAllySw = (__bf16*)carve((size_t)96 * H * 2);        // K padded 88->96
  __bf16* wEnSw   = (__bf16*)carve((size_t)96 * H * 2);        // K padded 80->96
  __bf16* wqaSw = (__bf16*)carve((size_t)H * H * 2);
  __bf16* wkaSw = (__bf16*)carve((size_t)H * H * 2);
  __bf16* wvaSw = (__bf16*)carve((size_t)H * H * 2);
  __bf16* woaSw = (__bf16*)carve((size_t)H * H * 2);
  __bf16* wqeSw = (__bf16*)carve((size_t)H * H * 2);
  __bf16* wkeSw = (__bf16*)carve((size_t)H * H * 2);
  __bf16* wveSw = (__bf16*)carve((size_t)H * H * 2);
  __bf16* woeSw = (__bf16*)carve((size_t)H * H * 2);
  __bf16* wihSw = (__bf16*)carve((size_t)3 * H * 3 * H * 2);   // K=1536,N=1536
  __bf16* whhSw = (__bf16*)carve((size_t)H * 3 * H * 2);       // K=512, N=1536
  float*  maskA = (float*)carve((size_t)B * 16 * 4);
  float*  maskE = (float*)carve((size_t)B * 16 * 4);
  __bf16* xbuf  = (__bf16*)carve((size_t)B * 3 * H * 2);       // [e_own|zA|zE]
  float*  qa    = (float*)carve((size_t)B * H * 4);
  float*  qe    = (float*)carve((size_t)B * H * 4);
  __bf16* atA   = (__bf16*)carve((size_t)B * H * 2);
  __bf16* atE   = (__bf16*)carve((size_t)B * H * 2);
  float*  gi    = (float*)carve((size_t)B * 3 * H * 4);
  float*  gh    = (float*)carve((size_t)B * 3 * H * 4);
  float*  hbuf  = (float*)carve((size_t)B * H * 4);

  // --- conversions / padding / swizzles ----------------------------------
  auto conv = [&](const float* src, __bf16* dst, long n) {
    long blocks = (n + 255) / 256;
    if (blocks > 4096) blocks = 4096;
    f2bf_kernel<<<dim3((unsigned)blocks), dim3(256), 0, stream>>>(src, dst, n);
  };
  conv(own, own16, (long)B * DO);
  conv(prevh, prevh16, (long)B * H);

  {
    long tot = (long)B * 16 * 96;
    unsigned blocks = (unsigned)((tot + 255) / 256);
    pad_ent_kernel<NA, DA><<<dim3(blocks), dim3(256), 0, stream>>>(ally, allyP, B);
    pad_ent_kernel<NE, DE><<<dim3(blocks), dim3(256), 0, stream>>>(enemy, enemyP, B);
  }

  auto swz = [&](const float* src, __bf16* dst, int K, int Ksrc, int N, int trans) {
    long tot = (long)K * N;
    unsigned blocks = (unsigned)((tot + 255) / 256);
    swizzleB_kernel<<<dim3(blocks), dim3(256), 0, stream>>>(src, dst, K, Ksrc, N, trans);
  };
  swz(W_own, wOwnSw, 96, DO, H, 0);
  swz(W_ally, wAllySw, 96, DA, H, 0);
  swz(W_en, wEnSw, 96, DE, H, 0);
  swz(Wq_a, wqaSw, H, H, H, 0);
  swz(Wk_a, wkaSw, H, H, H, 0);
  swz(Wv_a, wvaSw, H, H, H, 0);
  swz(Wo_a, woaSw, H, H, H, 0);
  swz(Wq_e, wqeSw, H, H, H, 0);
  swz(Wk_e, wkeSw, H, H, H, 0);
  swz(Wv_e, wveSw, H, H, H, 0);
  swz(Wo_e, woeSw, H, H, H, 0);
  swz(W_ih, wihSw, 3 * H, 3 * H, 3 * H, 1);  // gi = x @ W_ih^T
  swz(W_hh, whhSw, H, H, 3 * H, 1);          // gh = h @ W_hh^T

  mask_kernel<NA, DA><<<dim3(B * 16 / 256), dim3(256), 0, stream>>>(ally, maskA, B);
  mask_kernel<NE, DE><<<dim3(B * 16 / 256), dim3(256), 0, stream>>>(enemy, maskE, B);

  // --- GEMM helper (block tile 256x64) ------------------------------------
  auto gemm = [&](const __bf16* A, int lda, const __bf16* Bw, const float* bias,
                  float* Cf, __bf16* Cb, int ldc, int ccoff, int M, int N, int K) {
    dim3 grid((unsigned)(N / 64), (unsigned)(M / 256));
    gemm_bf16_kernel<<<grid, dim3(256), 0, stream>>>(A, lda, Bw, bias, Cf, Cb,
                                                     ldc, ccoff, M, N, K);
  };

  // e_own -> xbuf[:, 0:512]
  gemm(own16, DO, wOwnSw, b_own, nullptr, xbuf, 3 * H, 0, B, H, 96);
  // q_a = e_own @ Wq_a ; q_e = e_own @ Wq_e  (read xbuf cols 0..511, lda=1536)
  gemm(xbuf, 3 * H, wqaSw, nullptr, qa, nullptr, H, 0, B, H, H);
  gemm(xbuf, 3 * H, wqeSw, nullptr, qe, nullptr, H, 0, B, H, H);

  // fused attention (embedding + K/V projection + softmax + attn*V)
  attn_fused_kernel<<<dim3(B), dim3(256), 0, stream>>>(
      allyP, wAllySw, b_ally, wkaSw, wvaSw, qa, maskA, atA);
  attn_fused_kernel<<<dim3(B), dim3(256), 0, stream>>>(
      enemyP, wEnSw, b_en, wkeSw, wveSw, qe, maskE, atE);

  // zA -> xbuf[:, 512:1024] ; zE -> xbuf[:, 1024:1536]
  gemm(atA, H, woaSw, nullptr, nullptr, xbuf, 3 * H, H, B, H, H);
  gemm(atE, H, woeSw, nullptr, nullptr, xbuf, 3 * H, 2 * H, B, H, H);

  // gi = x @ W_ih^T + b_ih ; gh = prev_h @ W_hh^T + b_hh
  gemm(xbuf, 3 * H, wihSw, b_ih, gi, nullptr, 3 * H, 0, B, 3 * H, 3 * H);
  gemm(prevh16, H, whhSw, b_hh, gh, nullptr, 3 * H, 0, B, 3 * H, H);

  // GRU elementwise + Q head
  gru_kernel<<<dim3((B * H + 255) / 256), dim3(256), 0, stream>>>(gi, gh, prevh, hbuf, B);
  qout_kernel<<<dim3((B * 22 + 255) / 256), dim3(256), 0, stream>>>(hbuf, W_q, b_q, outq, B);
}
